// MambaLayer_58265526337686
// MI455X (gfx1250) — matrix operand order repaired and checked by reading, not compiled, not run on hardware
//
#include <hip/hip_runtime.h>
#include <hip/hip_bf16.h>

// ---------------------------------------------------------------------------
// Vision-Mamba layer for MI455X (gfx1250, wave32).
// GEMMs: bf16 V_WMMA_F32_16X16X32_BF16, one wave per 16x16 tile.
// Scan:  one lane per channel, 16 states in VGPRs, B/C via LDS broadcast.
// ---------------------------------------------------------------------------

#define B_SZ 2
#define DIM 256
#define L_SEQ 4096
#define D_INNER 512
#define D_STATE 16
#define DT_RANK 16
#define NS 64

typedef __attribute__((ext_vector_type(16))) __bf16 v16bf;
typedef __attribute__((ext_vector_type(8)))  __bf16 v8bf;
typedef __attribute__((ext_vector_type(8)))  float  v8f;
typedef __attribute__((ext_vector_type(4)))  unsigned int u32x4;

__device__ __forceinline__ unsigned short f2bf(float f) {
    unsigned int u = __builtin_bit_cast(unsigned int, f);
    unsigned int r = u + 0x7FFFu + ((u >> 16) & 1u);
    return (unsigned short)(r >> 16);
}

__device__ __forceinline__ float siluf(float x) {
    return x / (1.0f + __expf(-x));
}

// direction time map: 0 = identity, 1 = reverse, 2 = NS interleave (involution)
__device__ __forceinline__ int map_idx(int q, int dir) {
    if (dir == 0) return q;
    if (dir == 1) return (L_SEQ - 1) - q;
    return (q & (NS - 1)) * (L_SEQ / NS) + (q >> 6);  // (q%64)*64 + q/64
}

// ---------------------------------------------------------------------------
// f32 -> bf16 conversion with optional row padding:
//   dst[r*dcols + c] = (c < ccols) ? bf16(src[r*sstride + c]) : 0
// ---------------------------------------------------------------------------
__global__ __launch_bounds__(256)
void cvt_bf16_kernel(const float* __restrict__ src, unsigned short* __restrict__ dst,
                     int rows, int sstride, int ccols, int dcols) {
    int i = blockIdx.x * 256 + threadIdx.x;
    int total = rows * dcols;
    if (i >= total) return;
    int r = i / dcols, c = i - r * dcols;
    dst[i] = (c < ccols) ? f2bf(src[(size_t)r * sstride + c]) : (unsigned short)0;
}

// ---------------------------------------------------------------------------
// LayerNorm over channel dim; x is (B, C, L); writes bf16 (B*L, C) row-major.
// One block per (b,l) row, 256 threads = channels.
// ---------------------------------------------------------------------------
__global__ __launch_bounds__(256)
void ln_kernel(const float* __restrict__ x, const float* __restrict__ lnw,
               const float* __restrict__ lnb, unsigned short* __restrict__ xn_bf) {
    int row = blockIdx.x;                 // 0 .. B*L-1
    int b = row >> 12;                    // / 4096
    int l = row & (L_SEQ - 1);
    int c = threadIdx.x;
    float v = x[((size_t)b * DIM + c) * L_SEQ + l];
    __shared__ float sbuf[256];
    sbuf[c] = v; __syncthreads();
    for (int s = 128; s > 0; s >>= 1) { if (c < s) sbuf[c] += sbuf[c + s]; __syncthreads(); }
    float mu = sbuf[0] * (1.0f / 256.0f);
    __syncthreads();
    float d = v - mu;
    sbuf[c] = d * d; __syncthreads();
    for (int s = 128; s > 0; s >>= 1) { if (c < s) sbuf[c] += sbuf[c + s]; __syncthreads(); }
    float var = sbuf[0] * (1.0f / 256.0f);
    float xn = d * rsqrtf(var + 1e-5f) * lnw[c] + lnb[c];
    xn_bf[(size_t)row * DIM + c] = f2bf(xn);
}

// ---------------------------------------------------------------------------
// WMMA GEMM: C[M,N] = A[M,K] (bf16 rowmajor) * W[N,K]^T (bf16 rowmajor).
// One wave -> one 16x16 tile; K multiple of 32; M,N multiples of 16.
// mode 0: plain f32 store.  mode 1: softplus(acc + bias[n]).
// ---------------------------------------------------------------------------
__global__ __launch_bounds__(256)
void gemm_wmma_kernel(const unsigned short* __restrict__ A,
                      const unsigned short* __restrict__ W,
                      const float* __restrict__ bias,
                      float* __restrict__ C,
                      int M, int N, int K, int mode) {
    int lane = threadIdx.x & 31;
    int wave = threadIdx.x >> 5;
    int tiles_n = N >> 4;
    int tile = blockIdx.x * 8 + wave;
    int total = (M >> 4) * tiles_n;
    if (tile >= total) return;                       // whole-wave exit: EXEC stays all-1s
    int tm = tile / tiles_n, tn = tile - tm * tiles_n;

    int lrow  = lane & 15;
    int khalf = lane >> 4;
    int row = (tm << 4) + lrow;
    int col = (tn << 4) + lrow;
    const unsigned short* ap = A + (size_t)row * K + khalf * 8;    // A frag base
    const unsigned short* bp = W + (size_t)col * K + khalf * 8;    // B frag base

    v8f acc = {};
    for (int k0 = 0; k0 < K; k0 += 32) {
        __builtin_prefetch((const void*)(ap + k0 + 64), 0, 1);
        __builtin_prefetch((const void*)(bp + k0 + 64), 0, 1);
        u32x4 ra0 = *reinterpret_cast<const u32x4*>(ap + k0);        // K khalf*8 .. +7
        u32x4 ra1 = *reinterpret_cast<const u32x4*>(ap + k0 + 16);   // K 16+khalf*8 .. +7
        u32x4 rb0 = *reinterpret_cast<const u32x4*>(bp + k0);        // K khalf*8 .. +7
        u32x4 rb1 = *reinterpret_cast<const u32x4*>(bp + k0 + 16);   // K 16+khalf*8 .. +7
        v8bf alo = __builtin_bit_cast(v8bf, ra0);
        v8bf ahi = __builtin_bit_cast(v8bf, ra1);
        v8bf blo = __builtin_bit_cast(v8bf, rb0);
        v8bf bhi = __builtin_bit_cast(v8bf, rb1);
        v16bf av, bv;
#pragma unroll
        for (int i = 0; i < 8; i++) {
            av[i] = alo[i]; av[i + 8] = ahi[i];
            bv[i] = blo[i]; bv[i + 8] = bhi[i];
        }
        acc = __builtin_amdgcn_wmma_f32_16x16x32_bf16(
                  false, av, false, bv, (short)0, acc, false, false);
    }

    // D layout: VGPR r -> M = tm*16 + khalf*8 + r,  N = tn*16 + (lane&15)
    float* cp = C + (size_t)((tm << 4) + (khalf << 3)) * N + col;
    float bval = (mode == 1) ? bias[col] : 0.0f;
#pragma unroll
    for (int r = 0; r < 8; r++) {
        float v = acc[r];
        if (mode == 1) {
            v += bval;
            v = (v > 20.0f) ? v : logf(1.0f + __expf(v));   // softplus
        }
        cp[(size_t)r * N] = v;
    }
}

// ---------------------------------------------------------------------------
// Depthwise causal conv (K=4) on the direction-mapped sequence + SiLU.
// Reads xin = xz[:, :, 0:512]; writes f32 xc and bf16 xc for the next GEMM.
// ---------------------------------------------------------------------------
__global__ __launch_bounds__(256)
void conv_silu_kernel(const float* __restrict__ xz, const float* __restrict__ cw,
                      const float* __restrict__ cb, float* __restrict__ xc,
                      unsigned short* __restrict__ xc_bf, int dir) {
    int i = blockIdx.x * 256 + threadIdx.x;
    if (i >= B_SZ * L_SEQ * D_INNER) return;
    int d = i & (D_INNER - 1);
    int p = (i >> 9) & (L_SEQ - 1);
    int b = i >> 21;
    float acc = cb[d];
#pragma unroll
    for (int k = 0; k < 4; k++) {
        int q = p - 3 + k;
        if (q >= 0) {
            int t = map_idx(q, dir);
            acc += xz[((size_t)b * L_SEQ + t) * (2 * D_INNER) + d] * cw[d * 4 + k];
        }
    }
    float s = siluf(acc);
    xc[i] = s;
    xc_bf[i] = f2bf(s);
}

// ---------------------------------------------------------------------------
// Selective scan. One lane per channel; 16 states in registers.
// grid (D_INNER/256, B), block 256. B_t/C_t broadcast through LDS.
// Epilogue: +xc*D, gate with silu(z), scatter to canonical time, accumulate.
// ---------------------------------------------------------------------------
__global__ __launch_bounds__(256)
void scan_kernel(const float* __restrict__ dt, const float* __restrict__ xc,
                 const float* __restrict__ xdbl, const float* __restrict__ xz,
                 const float* __restrict__ A_log, const float* __restrict__ Dp,
                 float* __restrict__ ysum, int dir, int accum) {
    int b = blockIdx.y;
    int d = blockIdx.x * 256 + threadIdx.x;   // 0..511
    float Avals[D_STATE], h[D_STATE];
#pragma unroll
    for (int s = 0; s < D_STATE; s++) {
        Avals[s] = -__expf(A_log[d * D_STATE + s]);
        h[s] = 0.0f;
    }
    float Dv = Dp[d];
    __shared__ float sbc[2 * D_STATE];
    for (int t = 0; t < L_SEQ; t++) {
        if (threadIdx.x < 2 * D_STATE)
            sbc[threadIdx.x] = xdbl[((size_t)b * L_SEQ + t) * 48 + DT_RANK + threadIdx.x];
        __syncthreads();
        size_t base = ((size_t)b * L_SEQ + t) * D_INNER + d;
        float dtv = dt[base];
        float xv  = xc[base];
        float dx  = dtv * xv;
        float y = 0.0f;
#pragma unroll
        for (int s = 0; s < D_STATE; s++) {
            float dA = __expf(dtv * Avals[s]);
            h[s] = dA * h[s] + dx * sbc[s];
            y += h[s] * sbc[D_STATE + s];
        }
        y += xv * Dv;
        int tc = map_idx(t, dir);
        float zv = xz[((size_t)b * L_SEQ + tc) * (2 * D_INNER) + D_INNER + d];
        float out = y * siluf(zv);
        size_t ob = ((size_t)b * L_SEQ + tc) * D_INNER + d;
        if (accum) ysum[ob] += out; else ysum[ob] = out;
        __syncthreads();
    }
}

// ---------------------------------------------------------------------------
// out[b,c,h,w] = x[b,c,h,w] + proj[(b*L + l)*DIM + c]
// ---------------------------------------------------------------------------
__global__ __launch_bounds__(256)
void residual_kernel(const float* __restrict__ x, const float* __restrict__ proj,
                     float* __restrict__ out) {
    int i = blockIdx.x * 256 + threadIdx.x;
    if (i >= B_SZ * DIM * L_SEQ) return;
    int l = i & (L_SEQ - 1);
    int c = (i >> 12) & (DIM - 1);
    int b = i >> 20;
    out[i] = x[i] + proj[((size_t)b * L_SEQ + l) * DIM + c];
}

// ---------------------------------------------------------------------------
extern "C" void kernel_launch(void* const* d_in, const int* in_sizes, int n_in,
                              void* d_out, int out_size, void* d_ws, size_t ws_size,
                              hipStream_t stream) {
    (void)in_sizes; (void)n_in; (void)out_size; (void)ws_size;
    const float* x         = (const float*)d_in[0];
    const float* ln_w      = (const float*)d_in[1];
    const float* ln_b      = (const float*)d_in[2];
    const float* in_proj_w = (const float*)d_in[3];
    const float* conv_w[3]   = {(const float*)d_in[4],  (const float*)d_in[10], (const float*)d_in[16]};
    const float* conv_b[3]   = {(const float*)d_in[5],  (const float*)d_in[11], (const float*)d_in[17]};
    const float* x_proj_w[3] = {(const float*)d_in[6],  (const float*)d_in[12], (const float*)d_in[18]};
    const float* dt_proj_w[3]= {(const float*)d_in[7],  (const float*)d_in[13], (const float*)d_in[19]};
    const float* dt_proj_b[3]= {(const float*)d_in[8],  (const float*)d_in[14], (const float*)d_in[20]};
    const float* Dp[3]       = {(const float*)d_in[9],  (const float*)d_in[15], (const float*)d_in[21]};
    const float* A_log[3]    = {(const float*)d_in[22], (const float*)d_in[23], (const float*)d_in[24]};
    const float* out_proj_w  = (const float*)d_in[25];
    float* out = (float*)d_out;

    const int M = B_SZ * L_SEQ;   // 8192

    // workspace carve-up (256B aligned)
    char* ws = (char*)d_ws;
    size_t off = 0;
    auto take = [&](size_t bytes) -> char* {
        char* p = ws + off;
        off += (bytes + 255) & ~(size_t)255;
        return p;
    };
    unsigned short* xn_bf   = (unsigned short*)take((size_t)M * DIM * 2);
    unsigned short* w_in    = (unsigned short*)take((size_t)2 * D_INNER * DIM * 2);
    unsigned short* w_xp[3], *w_dt[3];
    for (int i = 0; i < 3; i++) w_xp[i] = (unsigned short*)take((size_t)48 * D_INNER * 2);
    for (int i = 0; i < 3; i++) w_dt[i] = (unsigned short*)take((size_t)D_INNER * 32 * 2);
    unsigned short* w_out   = (unsigned short*)take((size_t)DIM * D_INNER * 2);
    float*          xz      = (float*)take((size_t)M * 2 * D_INNER * 4);
    float*          xc      = (float*)take((size_t)M * D_INNER * 4);
    unsigned short* xc_bf   = (unsigned short*)take((size_t)M * D_INNER * 2);
    float*          xdbl    = (float*)take((size_t)M * 48 * 4);
    unsigned short* dtr_bf  = (unsigned short*)take((size_t)M * 32 * 2);
    float*          dtbuf   = (float*)take((size_t)M * D_INNER * 4);
    float*          ysum    = (float*)take((size_t)M * D_INNER * 4);
    unsigned short* ysum_bf = (unsigned short*)take((size_t)M * D_INNER * 2);
    float*          proj    = (float*)take((size_t)M * DIM * 4);

    auto cvt = [&](const float* s, unsigned short* d, int rows, int sstride, int ccols, int dcols) {
        int total = rows * dcols;
        cvt_bf16_kernel<<<(total + 255) / 256, 256, 0, stream>>>(s, d, rows, sstride, ccols, dcols);
    };

    // weights -> bf16 (dt_proj padded K 16->32)
    cvt(in_proj_w, w_in, 2 * D_INNER, DIM, DIM, DIM);
    for (int i = 0; i < 3; i++) cvt(x_proj_w[i], w_xp[i], 48, D_INNER, D_INNER, D_INNER);
    for (int i = 0; i < 3; i++) cvt(dt_proj_w[i], w_dt[i], D_INNER, DT_RANK, DT_RANK, 32);
    cvt(out_proj_w, w_out, DIM, D_INNER, D_INNER, D_INNER);

    // layernorm -> bf16
    ln_kernel<<<M, 256, 0, stream>>>(x, ln_w, ln_b, xn_bf);

    // in_proj: (M x 256) @ (1024 x 256)^T -> xz f32
    {
        int tiles = (M / 16) * (2 * D_INNER / 16);
        gemm_wmma_kernel<<<tiles / 8, 256, 0, stream>>>(xn_bf, w_in, nullptr, xz,
                                                        M, 2 * D_INNER, DIM, 0);
    }

    for (int dir = 0; dir < 3; dir++) {
        // causal conv + silu on mapped sequence
        conv_silu_kernel<<<(B_SZ * L_SEQ * D_INNER + 255) / 256, 256, 0, stream>>>(
            xz, conv_w[dir], conv_b[dir], xc, xc_bf, dir);
        // x_proj: (M x 512) @ (48 x 512)^T -> xdbl
        {
            int tiles = (M / 16) * (48 / 16);
            gemm_wmma_kernel<<<tiles / 8, 256, 0, stream>>>(xc_bf, w_xp[dir], nullptr, xdbl,
                                                            M, 48, D_INNER, 0);
        }
        // dt_r (cols 0..15 of xdbl) -> bf16, K padded to 32
        cvt(xdbl, dtr_bf, M, 48, DT_RANK, 32);
        // dt_proj + softplus: (M x 32) @ (512 x 32)^T
        {
            int tiles = (M / 16) * (D_INNER / 16);
            gemm_wmma_kernel<<<tiles / 8, 256, 0, stream>>>(dtr_bf, w_dt[dir], dt_proj_b[dir],
                                                            dtbuf, M, D_INNER, 32, 1);
        }
        // selective scan + gate + scatter-accumulate into ysum
        scan_kernel<<<dim3(D_INNER / 256, B_SZ), 256, 0, stream>>>(
            dtbuf, xc, xdbl, xz, A_log[dir], Dp[dir], ysum, dir, dir > 0 ? 1 : 0);
    }

    // ysum -> bf16, out_proj, residual
    cvt(ysum, ysum_bf, M, D_INNER, D_INNER, D_INNER);
    {
        int tiles = (M / 16) * (DIM / 16);
        gemm_wmma_kernel<<<tiles / 8, 256, 0, stream>>>(ysum_bf, w_out, nullptr, proj,
                                                        M, DIM, D_INNER, 0);
    }
    residual_kernel<<<(B_SZ * DIM * L_SEQ + 255) / 256, 256, 0, stream>>>(x, proj, out);
}
